// ModernBERTFlashAttnBlock_85529978733225
// MI455X (gfx1250) — matrix operand order repaired and checked by reading, not compiled, not run on hardware
//
#include <hip/hip_runtime.h>
#include <hip/hip_bf16.h>
#include <math.h>

#define B_   2
#define M_   2048
#define D_   1024
#define H_   16
#define DH_  64
#define FF_  2624
#define BM_  (B_ * M_)
#define D3_  (3 * D_)
#define FF2_ (2 * FF_)
#define EPS_ 1e-5f

typedef _Float16 f16;
typedef _Float16 v16h __attribute__((ext_vector_type(16)));
typedef _Float16 v8h  __attribute__((ext_vector_type(8)));
typedef float    v8f  __attribute__((ext_vector_type(8)));

__device__ __forceinline__ v8f wmma_f16(v16h a, v16h b, v8f c) {
  // D = A(16x32 f16) x B(32x16 f16) + C(16x16 f32)
  return __builtin_amdgcn_wmma_f32_16x16x32_f16(false, a, false, b, (short)0, c,
                                                false, false);
}

// A fragment (16x32 f16), row-major source [*, lda]:
// lane l -> row m0+(l&15); halves[0..15]=k0..k0+15
// with k0 = kb + (l>>4)*16   (contiguous K order, matches load_b_frag)
__device__ __forceinline__ v16h load_a_frag(const f16* __restrict__ A, int lda,
                                            int m0, int kb, int lane) {
  int m  = m0 + (lane & 15);
  int k0 = kb + ((lane >> 4) << 4);
  v8h lo = *(const v8h*)(A + m * lda + k0);
  v8h hi = *(const v8h*)(A + m * lda + k0 + 8);
  v16h a;
#pragma unroll
  for (int i = 0; i < 8; ++i) { a[i] = lo[i]; a[i + 8] = hi[i]; }
  return a;
}

// B fragment (32x16 f16) where B[k][n] = W[n][k], W row-major [N,K]:
// lane l -> col n0+(l&15); 16 contiguous K halves starting at kb+(l>>4)*16
__device__ __forceinline__ v16h load_b_frag(const f16* __restrict__ W, int ldb,
                                            int n0, int kb, int lane) {
  int n  = n0 + (lane & 15);
  int k0 = kb + ((lane >> 4) << 4);
  return *(const v16h*)(W + n * ldb + k0);
}

// ---------------------------------------------------------------- weights f32->f16
__global__ void cvt_f32_to_f16(const float* __restrict__ src, f16* __restrict__ dst,
                               int n) {
  int i = blockIdx.x * blockDim.x + threadIdx.x;
  if (i < n) dst[i] = (f16)src[i];
}

// ---------------------------------------------------------------- LayerNorm -> f16
__global__ void __launch_bounds__(256)
layernorm_to_f16(const float* __restrict__ X, const float* __restrict__ g,
                 const float* __restrict__ bb, f16* __restrict__ Y) {
  int row = blockIdx.x;
  int tid = threadIdx.x;
  const float4 v = ((const float4*)(X + row * D_))[tid];
  float s  = v.x + v.y + v.z + v.w;
  float s2 = v.x * v.x + v.y * v.y + v.z * v.z + v.w * v.w;
#pragma unroll
  for (int o = 16; o > 0; o >>= 1) {
    s  += __shfl_xor(s, o, 32);
    s2 += __shfl_xor(s2, o, 32);
  }
  __shared__ float sh[16];
  if ((tid & 31) == 0) { sh[tid >> 5] = s; sh[8 + (tid >> 5)] = s2; }
  __syncthreads();
  float ts = 0.f, ts2 = 0.f;
#pragma unroll
  for (int i = 0; i < 8; ++i) { ts += sh[i]; ts2 += sh[8 + i]; }
  float mu  = ts * (1.0f / D_);
  float var = ts2 * (1.0f / D_) - mu * mu;
  float inv = rsqrtf(var + EPS_);
  int c = tid * 4;
  float4 gv = ((const float4*)g)[tid];
  float4 bv = ((const float4*)bb)[tid];
  f16* yp = Y + row * D_ + c;
  yp[0] = (f16)((v.x - mu) * inv * gv.x + bv.x);
  yp[1] = (f16)((v.y - mu) * inv * gv.y + bv.y);
  yp[2] = (f16)((v.z - mu) * inv * gv.z + bv.z);
  yp[3] = (f16)((v.w - mu) * inv * gv.w + bv.w);
}

// ------------------------------------------------- GEMM: C=f16(A@W^T + bias) (QKV)
// Software-pipelined: next iteration's fragments loaded before current WMMAs.
__global__ void __launch_bounds__(128)
gemm_bias_out_f16(const f16* __restrict__ A, const f16* __restrict__ Bw,
                  const float* __restrict__ bias, f16* __restrict__ C, int K, int N) {
  int lane = threadIdx.x & 31;
  int wave = threadIdx.x >> 5;
  int m0 = blockIdx.y * 64 + wave * 16;
  int n0 = blockIdx.x * 64;
  v8f acc0 = {}, acc1 = {}, acc2 = {}, acc3 = {};
  v16h a  = load_a_frag(A, K, m0, 0, lane);
  v16h b0 = load_b_frag(Bw, K, n0 +  0, 0, lane);
  v16h b1 = load_b_frag(Bw, K, n0 + 16, 0, lane);
  v16h b2 = load_b_frag(Bw, K, n0 + 32, 0, lane);
  v16h b3 = load_b_frag(Bw, K, n0 + 48, 0, lane);
  for (int kb = 0; kb < K; kb += 32) {
    int kn = (kb + 32 < K) ? kb + 32 : 0;   // branchless wrap on last iter
    v16h an = load_a_frag(A, K, m0, kn, lane);
    v16h c0 = load_b_frag(Bw, K, n0 +  0, kn, lane);
    v16h c1 = load_b_frag(Bw, K, n0 + 16, kn, lane);
    v16h c2 = load_b_frag(Bw, K, n0 + 32, kn, lane);
    v16h c3 = load_b_frag(Bw, K, n0 + 48, kn, lane);
    __builtin_prefetch(A + (m0 + (lane & 15)) * K + kb + 128, 0, 3);
    acc0 = wmma_f16(a, b0, acc0);
    acc1 = wmma_f16(a, b1, acc1);
    acc2 = wmma_f16(a, b2, acc2);
    acc3 = wmma_f16(a, b3, acc3);
    a = an; b0 = c0; b1 = c1; b2 = c2; b3 = c3;
  }
  int mr = m0 + ((lane >> 4) << 3);
  int nc = n0 + (lane & 15);
  v8f acc[4] = {acc0, acc1, acc2, acc3};
#pragma unroll
  for (int t = 0; t < 4; ++t) {
    int col = nc + t * 16;
    float bcol = bias[col];
#pragma unroll
    for (int r = 0; r < 8; ++r)
      C[(mr + r) * N + col] = (f16)(acc[t][r] + bcol);
  }
}

// ------------------------------------------ GEMM: C=f32(A@W^T + resid) (Wo, MLP-out)
__global__ void __launch_bounds__(128)
gemm_resid_out_f32(const f16* __restrict__ A, const f16* __restrict__ Bw,
                   const float* __restrict__ resid, float* __restrict__ C, int K,
                   int N) {
  int lane = threadIdx.x & 31;
  int wave = threadIdx.x >> 5;
  int m0 = blockIdx.y * 64 + wave * 16;
  int n0 = blockIdx.x * 64;
  v8f acc0 = {}, acc1 = {}, acc2 = {}, acc3 = {};
  v16h a  = load_a_frag(A, K, m0, 0, lane);
  v16h b0 = load_b_frag(Bw, K, n0 +  0, 0, lane);
  v16h b1 = load_b_frag(Bw, K, n0 + 16, 0, lane);
  v16h b2 = load_b_frag(Bw, K, n0 + 32, 0, lane);
  v16h b3 = load_b_frag(Bw, K, n0 + 48, 0, lane);
  for (int kb = 0; kb < K; kb += 32) {
    int kn = (kb + 32 < K) ? kb + 32 : 0;
    v16h an = load_a_frag(A, K, m0, kn, lane);
    v16h c0 = load_b_frag(Bw, K, n0 +  0, kn, lane);
    v16h c1 = load_b_frag(Bw, K, n0 + 16, kn, lane);
    v16h c2 = load_b_frag(Bw, K, n0 + 32, kn, lane);
    v16h c3 = load_b_frag(Bw, K, n0 + 48, kn, lane);
    __builtin_prefetch(A + (m0 + (lane & 15)) * K + kb + 128, 0, 3);
    acc0 = wmma_f16(a, b0, acc0);
    acc1 = wmma_f16(a, b1, acc1);
    acc2 = wmma_f16(a, b2, acc2);
    acc3 = wmma_f16(a, b3, acc3);
    a = an; b0 = c0; b1 = c1; b2 = c2; b3 = c3;
  }
  int mr = m0 + ((lane >> 4) << 3);
  int nc = n0 + (lane & 15);
  v8f acc[4] = {acc0, acc1, acc2, acc3};
#pragma unroll
  for (int t = 0; t < 4; ++t) {
    int col = nc + t * 16;
#pragma unroll
    for (int r = 0; r < 8; ++r) {
      int idx = (mr + r) * N + col;
      C[idx] = acc[t][r] + resid[idx];
    }
  }
}

// --------------------------- fused Wi GEMM + gelu(h_in)*h_gate epilogue -> act f16
// A double-buffered; the 8 B loads of an iteration are batched ahead of its WMMAs.
__global__ void __launch_bounds__(128)
gemm_mlp_gelu(const f16* __restrict__ A, const f16* __restrict__ Wi,
              f16* __restrict__ act) {
  const int K = D_;
  int lane = threadIdx.x & 31;
  int wave = threadIdx.x >> 5;
  int m0 = blockIdx.y * 64 + wave * 16;
  int n0 = blockIdx.x * 64;
  v8f ai[4] = {}, ag[4] = {};
  v16h a = load_a_frag(A, K, m0, 0, lane);
  for (int kb = 0; kb < K; kb += 32) {
    int kn = (kb + 32 < K) ? kb + 32 : 0;
    v16h an = load_a_frag(A, K, m0, kn, lane);
    v16h bi[4], bg[4];
#pragma unroll
    for (int t = 0; t < 4; ++t) {
      bi[t] = load_b_frag(Wi, K, n0 + t * 16, kb, lane);
      bg[t] = load_b_frag(Wi, K, FF_ + n0 + t * 16, kb, lane);
    }
#pragma unroll
    for (int t = 0; t < 4; ++t) {
      ai[t] = wmma_f16(a, bi[t], ai[t]);
      ag[t] = wmma_f16(a, bg[t], ag[t]);
    }
    a = an;
  }
  int mr = m0 + ((lane >> 4) << 3);
  int nc = n0 + (lane & 15);
#pragma unroll
  for (int t = 0; t < 4; ++t) {
#pragma unroll
    for (int r = 0; r < 8; ++r) {
      float xv = ai[t][r];
      float ge = 0.5f * xv *
                 (1.0f + tanhf(0.7978845608f * (xv + 0.044715f * xv * xv * xv)));
      act[(mr + r) * FF_ + nc + t * 16] = (f16)(ge * ag[t][r]);
    }
  }
}

// ---------------- RoPE + head split: qkv[BM,3D] -> q,k [B,H,M,DH] (q pre-scaled),
// ---------------- v transposed to [B,H,DH,M] for contiguous P.V B-fragments
__global__ void __launch_bounds__(256)
rope_split(const f16* __restrict__ qkv, f16* __restrict__ q, f16* __restrict__ k,
           f16* __restrict__ vt) {
  int idx = blockIdx.x * blockDim.x + threadIdx.x; // B*M*H*32 threads
  int d2 = idx & 31; idx >>= 5;
  int h  = idx & 15; idx >>= 4;
  int m  = idx & (M_ - 1); idx >>= 11;
  int b  = idx;
  int row = b * M_ + m;
  const f16* base = qkv + row * D3_;
  float invf = __expf(-(float)(2 * d2) * (1.0f / 64.0f) * 9.2103403720f); // ln(1e4)
  float ang = (float)m * invf;
  float c, s;
  __sincosf(ang, &s, &c);
  float q1 = (float)base[0 * D_ + h * 64 + d2];
  float q2 = (float)base[0 * D_ + h * 64 + d2 + 32];
  float k1 = (float)base[1 * D_ + h * 64 + d2];
  float k2 = (float)base[1 * D_ + h * 64 + d2 + 32];
  float v1 = (float)base[2 * D_ + h * 64 + d2];
  float v2 = (float)base[2 * D_ + h * 64 + d2 + 32];
  const float scale = 0.125f; // 1/sqrt(DH)
  size_t o = ((size_t)(b * H_ + h) * M_ + m) * DH_;
  q[o + d2]      = (f16)((q1 * c - q2 * s) * scale);
  q[o + d2 + 32] = (f16)((q1 * s + q2 * c) * scale);
  k[o + d2]      = (f16)(k1 * c - k2 * s);
  k[o + d2 + 32] = (f16)(k1 * s + k2 * c);
  size_t ov = (size_t)(b * H_ + h) * DH_ * M_;
  vt[ov + (size_t)d2 * M_ + m]        = (f16)v1;
  vt[ov + (size_t)(d2 + 32) * M_ + m] = (f16)v2;
}

// --------------------------------- flash attention: one wave per 16-row Q tile
// K fragments double-buffered across j; V fragments issued at top of body so the
// softmax VALU phase covers their latency.
__global__ void __launch_bounds__(32)
flash_attn(const f16* __restrict__ q, const f16* __restrict__ k,
           const f16* __restrict__ vt, const int* __restrict__ mask,
           f16* __restrict__ attn_out) {
  int lane = threadIdx.x;
  int mt = blockIdx.x;            // query tile 0..M/16-1
  int bh = blockIdx.y;            // 0..B*H-1
  int b = bh >> 4, h = bh & 15;
  const f16* qh = q  + (size_t)bh * M_ * DH_;
  const f16* kh = k  + (size_t)bh * M_ * DH_;
  const f16* vh = vt + (size_t)bh * DH_ * M_;
  const int* mrow = mask + b * M_;
  int m0 = mt * 16;

  v16h aq0 = load_a_frag(qh, DH_, m0, 0, lane);   // Q d=0..31 (pre-scaled)
  v16h aq1 = load_a_frag(qh, DH_, m0, 32, lane);  // Q d=32..63
  v8f o0 = {}, o1 = {}, o2 = {}, o3 = {};
  float rmax[8], rsum[8];
#pragma unroll
  for (int r = 0; r < 8; ++r) { rmax[r] = -1e30f; rsum[r] = 0.f; }

  __shared__ f16 lp[16 * 32]; // P tile staging for C-layout -> A-layout

  int nL = lane & 15;
  int gHi = (lane >> 4) << 3;   // row offset for C layout (0 or 8)

  // preload K fragments for j = 0
  v16h bk0 = load_b_frag(kh, DH_,  0,  0, lane);
  v16h bk1 = load_b_frag(kh, DH_,  0, 32, lane);
  v16h bk2 = load_b_frag(kh, DH_, 16,  0, lane);
  v16h bk3 = load_b_frag(kh, DH_, 16, 32, lane);

  for (int j = 0; j < M_; j += 32) {
    int jn = (j + 32 < M_) ? j + 32 : 0;  // branchless wrap on last iter
    // ---- issue current V loads early (consumed after softmax)
    int kc = j + ((lane >> 4) << 4);
    v16h bv0 = *(const v16h*)(vh + (size_t)(0 * 16 + nL) * M_ + kc);
    v16h bv1 = *(const v16h*)(vh + (size_t)(1 * 16 + nL) * M_ + kc);
    v16h bv2 = *(const v16h*)(vh + (size_t)(2 * 16 + nL) * M_ + kc);
    v16h bv3 = *(const v16h*)(vh + (size_t)(3 * 16 + nL) * M_ + kc);
    // ---- issue next-iteration K loads early
    v16h nk0 = load_b_frag(kh, DH_, jn,      0, lane);
    v16h nk1 = load_b_frag(kh, DH_, jn,     32, lane);
    v16h nk2 = load_b_frag(kh, DH_, jn + 16,  0, lane);
    v16h nk3 = load_b_frag(kh, DH_, jn + 16, 32, lane);
    // ---- S = Q K^T for 32 keys (two 16x16 tiles)
    v8f s0 = {}, s1 = {};
    s0 = wmma_f16(aq0, bk0, s0);
    s0 = wmma_f16(aq1, bk1, s0);
    s1 = wmma_f16(aq0, bk2, s1);
    s1 = wmma_f16(aq1, bk3, s1);
    float mk0 = mrow[j + nL]      ? 0.f : -1e30f;
    float mk1 = mrow[j + 16 + nL] ? 0.f : -1e30f;
    // ---- online softmax per row (row = gHi+r, spread over 16 lanes)
#pragma unroll
    for (int r = 0; r < 8; ++r) {
      float a0 = s0[r] + mk0, a1 = s1[r] + mk1;
      float mx = fmaxf(a0, a1);
#pragma unroll
      for (int o = 1; o < 16; o <<= 1) mx = fmaxf(mx, __shfl_xor(mx, o, 16));
      float nm = fmaxf(rmax[r], mx);
      float fs = __expf(rmax[r] - nm);
      rmax[r] = nm;
      float p0 = __expf(a0 - nm), p1 = __expf(a1 - nm);
      float ps = p0 + p1;
#pragma unroll
      for (int o = 1; o < 16; o <<= 1) ps += __shfl_xor(ps, o, 16);
      rsum[r] = rsum[r] * fs + ps;
      o0[r] *= fs; o1[r] *= fs; o2[r] *= fs; o3[r] *= fs;
      lp[(gHi + r) * 32 + nL]      = (f16)p0;
      lp[(gHi + r) * 32 + 16 + nL] = (f16)p1;
    }
    __syncthreads();
    // ---- reload P as A fragment (16 rows x 32 keys)
    v16h ap;
    {
      int mm = lane & 15;
      int k0 = (lane >> 4) << 4;
      v8h lo = *(const v8h*)(lp + mm * 32 + k0);
      v8h hi = *(const v8h*)(lp + mm * 32 + 8 + k0);
#pragma unroll
      for (int i = 0; i < 8; ++i) { ap[i] = lo[i]; ap[i + 8] = hi[i]; }
    }
    __syncthreads();
    // ---- O += P V  (V pre-transposed: contiguous along key dim)
    o0 = wmma_f16(ap, bv0, o0);
    o1 = wmma_f16(ap, bv1, o1);
    o2 = wmma_f16(ap, bv2, o2);
    o3 = wmma_f16(ap, bv3, o3);
    bk0 = nk0; bk1 = nk1; bk2 = nk2; bk3 = nk3;
  }
  // ---- normalize + write [B,M,D] f16 (head-interleaved)
#pragma unroll
  for (int r = 0; r < 8; ++r) {
    float inv = 1.0f / rsum[r];
    int row = m0 + gHi + r;
    size_t ob = ((size_t)b * M_ + row) * D_ + h * 64;
    attn_out[ob + 0  + nL] = (f16)(o0[r] * inv);
    attn_out[ob + 16 + nL] = (f16)(o1[r] * inv);
    attn_out[ob + 32 + nL] = (f16)(o2[r] * inv);
    attn_out[ob + 48 + nL] = (f16)(o3[r] * inv);
  }
}

extern "C" void kernel_launch(void* const* d_in, const int* in_sizes, int n_in,
                              void* d_out, int out_size, void* d_ws, size_t ws_size,
                              hipStream_t stream) {
  (void)in_sizes; (void)n_in; (void)out_size; (void)ws_size;
  const float* x      = (const float*)d_in[0];
  const int*   amask  = (const int*)d_in[1];
  const float* ln1_g  = (const float*)d_in[2];
  const float* ln1_b  = (const float*)d_in[3];
  const float* Wqkv_w = (const float*)d_in[4];
  const float* Wqkv_b = (const float*)d_in[5];
  const float* Wo_w   = (const float*)d_in[6];
  const float* ln2_g  = (const float*)d_in[7];
  const float* ln2_b  = (const float*)d_in[8];
  const float* Wi_w   = (const float*)d_in[9];
  const float* Wmo_w  = (const float*)d_in[10];
  float* out = (float*)d_out;

  char* ws = (char*)d_ws;
  size_t off = 0;
  auto alloc = [&](size_t bytes) -> void* {
    void* p = ws + off;
    off += (bytes + 255) & ~(size_t)255;
    return p;
  };
  f16*   xn   = (f16*)alloc((size_t)BM_ * D_ * 2);
  f16*   wqkv = (f16*)alloc((size_t)D3_ * D_ * 2);
  f16*   wo   = (f16*)alloc((size_t)D_ * D_ * 2);
  f16*   wi   = (f16*)alloc((size_t)FF2_ * D_ * 2);
  f16*   wmo  = (f16*)alloc((size_t)D_ * FF_ * 2);
  f16*   qkv  = (f16*)alloc((size_t)BM_ * D3_ * 2);
  f16*   qr   = (f16*)alloc((size_t)BM_ * D_ * 2);
  f16*   kr   = (f16*)alloc((size_t)BM_ * D_ * 2);
  f16*   vt   = (f16*)alloc((size_t)BM_ * D_ * 2);
  f16*   attn = (f16*)alloc((size_t)BM_ * D_ * 2);
  float* x1   = (float*)alloc((size_t)BM_ * D_ * 4);
  f16*   xn2  = (f16*)alloc((size_t)BM_ * D_ * 2);
  f16*   act  = (f16*)alloc((size_t)BM_ * FF_ * 2);
  // total workspace ~138 MB

  auto cvt = [&](const float* s, f16* dpt, int n) {
    cvt_f32_to_f16<<<(n + 255) / 256, 256, 0, stream>>>(s, dpt, n);
  };
  cvt(Wqkv_w, wqkv, D3_ * D_);
  cvt(Wo_w,   wo,   D_ * D_);
  cvt(Wi_w,   wi,   FF2_ * D_);
  cvt(Wmo_w,  wmo,  D_ * FF_);

  layernorm_to_f16<<<BM_, 256, 0, stream>>>(x, ln1_g, ln1_b, xn);
  gemm_bias_out_f16<<<dim3(D3_ / 64, BM_ / 64), 128, 0, stream>>>(
      xn, wqkv, Wqkv_b, qkv, D_, D3_);
  rope_split<<<(B_ * M_ * H_ * 32) / 256, 256, 0, stream>>>(qkv, qr, kr, vt);
  flash_attn<<<dim3(M_ / 16, B_ * H_), 32, 0, stream>>>(qr, kr, vt, amask, attn);
  gemm_resid_out_f32<<<dim3(D_ / 64, BM_ / 64), 128, 0, stream>>>(
      attn, wo, x, x1, D_, D_);
  layernorm_to_f16<<<BM_, 256, 0, stream>>>(x1, ln2_g, ln2_b, xn2);
  gemm_mlp_gelu<<<dim3(FF_ / 64, BM_ / 64), 128, 0, stream>>>(xn2, wi, act);
  gemm_resid_out_f32<<<dim3(D_ / 64, BM_ / 64), 128, 0, stream>>>(
      act, wmo, x1, out, FF_, D_);
}
